// LowRankLinearAttention_46737834115545
// MI455X (gfx1250) — matrix operand order repaired and checked by reading, not compiled, not run on hardware
//
#include <hip/hip_runtime.h>

// ---------------------------------------------------------------------------
// CDNA5 (gfx1250) implementation: low-rank linear attention + dense graph-bias
// GEMMs via v_wmma_f32_16x16x32_bf16 (wave32); dominant GEMM uses
// GLOBAL_LOAD_ASYNC_TO_LDS_B128 double-buffered staging (ASYNCcnt).
// ---------------------------------------------------------------------------

typedef __bf16 bf16;
typedef __attribute__((ext_vector_type(16))) __bf16 v16bf;
typedef __attribute__((ext_vector_type(8)))  __bf16 v8bf;
typedef __attribute__((ext_vector_type(8)))  float  v8f;

#define BB 8
#define NN 4096
#define DD 512
#define HH 8
#define RK 64
#define HD 64
#define ROWS (BB * NN)          // 32768
#define LDST 40                 // padded LDS row stride (halfs) for 32-wide tiles
#define LDT  72                 // padded LDS row stride (halfs) for 64-wide T tile
#define AM  128                 // rows per block in attn kernel

union Frag { v16bf v; v8bf h[2]; };

__device__ __forceinline__ v8f wmma_bf16(const Frag& a, const Frag& b, v8f c) {
    return __builtin_amdgcn_wmma_f32_16x16x32_bf16(
        false, a.v, false, b.v, (short)0, c, false, false);
}

// A-fragment: 16x32 bf16 tile, row-major in LDS.
// lane: m = lane&15, kgrp = lane>>4.
__device__ __forceinline__ Frag load_a(const bf16* rowptr, int kgrp) {
    Frag f;
    f.h[0] = *(const v8bf*)(rowptr + kgrp * 8);
    f.h[1] = *(const v8bf*)(rowptr + 16 + kgrp * 8);
    return f;
}
// B-fragment: 32x16 tile stored TRANSPOSED in LDS (Bt[n][k]).
// lane: n = lane&15, khalf = lane>>4.
__device__ __forceinline__ Frag load_b(const bf16* colptr, int khalf) {
    Frag f;
    f.h[0] = *(const v8bf*)(colptr + khalf * 8);
    f.h[1] = *(const v8bf*)(colptr + 16 + khalf * 8);
    return f;
}

// ---------------------------------------------------------------------------
// f32 -> bf16 convert
// ---------------------------------------------------------------------------
__global__ __launch_bounds__(256) void cvt_f32_bf16(const float* __restrict__ in,
                                                    bf16* __restrict__ out, int n) {
    int i = blockIdx.x * 256 + threadIdx.x;
    if (i < n) out[i] = (bf16)in[i];
}

// ---------------------------------------------------------------------------
// Row softmax of graph_bias (4096 x 4096 f32) -> bf16 probs
// ---------------------------------------------------------------------------
__global__ __launch_bounds__(256) void softmax_kernel(const float* __restrict__ gb,
                                                      bf16* __restrict__ P) {
    const int row = blockIdx.x;
    const int t = threadIdx.x;
    const float* src = gb + (size_t)row * NN;
    __shared__ float red[256];

    float vals[16];
    float m = -3.0e38f;
    #pragma unroll
    for (int j = 0; j < 16; ++j) {
        vals[j] = src[t + j * 256];
        m = fmaxf(m, vals[j]);
    }
    red[t] = m;
    __syncthreads();
    for (int s = 128; s > 0; s >>= 1) {
        if (t < s) red[t] = fmaxf(red[t], red[t + s]);
        __syncthreads();
    }
    const float rowmax = red[0];
    __syncthreads();

    float sum = 0.f;
    #pragma unroll
    for (int j = 0; j < 16; ++j) {
        vals[j] = __expf(vals[j] - rowmax);
        sum += vals[j];
    }
    red[t] = sum;
    __syncthreads();
    for (int s = 128; s > 0; s >>= 1) {
        if (t < s) red[t] += red[t + s];
        __syncthreads();
    }
    const float inv = 1.0f / red[0];
    bf16* dst = P + (size_t)row * NN;
    #pragma unroll
    for (int j = 0; j < 16; ++j) dst[t + j * 256] = (bf16)(vals[j] * inv);
}

// ---------------------------------------------------------------------------
// Low-rank projection:  Y = f( (in @ Wd^T) @ Wu^T )
// mode 0: Y bf16,  mode 1: Y bf16 with elu+1,  mode 2: Y f32 plain
// ---------------------------------------------------------------------------
__global__ __launch_bounds__(256) void proj_kernel(const bf16* __restrict__ in,
                                                   const bf16* __restrict__ wdh,
                                                   const bf16* __restrict__ wuh,
                                                   void* __restrict__ outp, int mode) {
    __shared__ __align__(16) bf16 Alds[64 * LDST];
    __shared__ __align__(16) bf16 Btlds[64 * LDST];
    __shared__ __align__(16) bf16 Tlds[64 * LDT];

    const int tid  = threadIdx.x;
    const int lane = tid & 31;
    const int wave = tid >> 5;
    const int mi   = wave >> 1;
    const int ni0  = (wave & 1) * 2;
    const int ni1  = ni0 + 1;
    const int mfr  = lane & 15;
    const int kgrp = lane >> 4;
    const size_t row0 = (size_t)blockIdx.x * 64;

    // ---------------- stage 1: T(64x64) = X(64x512) @ Wd^T ----------------
    v8f t0 = {}, t1 = {};
    const int lm = tid >> 2;
    const int lk = (tid & 3) * 8;
    for (int ks = 0; ks < 16; ++ks) {
        __syncthreads();
        *(v8bf*)(Alds + lm * LDST + lk) =
            *(const v8bf*)(in + (row0 + lm) * DD + ks * 32 + lk);
        *(v8bf*)(Btlds + lm * LDST + lk) =
            *(const v8bf*)(wdh + (size_t)lm * DD + ks * 32 + lk);
        __syncthreads();
        Frag a  = load_a(Alds + (mi * 16 + mfr) * LDST, kgrp);
        Frag b0 = load_b(Btlds + (ni0 * 16 + mfr) * LDST, kgrp);
        Frag b1 = load_b(Btlds + (ni1 * 16 + mfr) * LDST, kgrp);
        t0 = wmma_bf16(a, b0, t0);
        t1 = wmma_bf16(a, b1, t1);
    }
    __syncthreads();
    #pragma unroll
    for (int r = 0; r < 8; ++r) {
        int mr = mi * 16 + r + (kgrp << 3);
        Tlds[mr * LDT + ni0 * 16 + mfr] = (bf16)t0[r];
        Tlds[mr * LDT + ni1 * 16 + mfr] = (bf16)t1[r];
    }

    // ------------- stage 2: Y(64x512) = T(64x64) @ Wu^T -------------
    for (int cg = 0; cg < 8; ++cg) {
        v8f c0 = {}, c1 = {};
        for (int ks2 = 0; ks2 < 2; ++ks2) {
            __syncthreads();
            *(v8bf*)(Btlds + lm * LDST + lk) =
                *(const v8bf*)(wuh + (size_t)(cg * 64 + lm) * RK + ks2 * 32 + lk);
            __syncthreads();
            Frag a  = load_a(Tlds + (mi * 16 + mfr) * LDT + ks2 * 32, kgrp);
            Frag b0 = load_b(Btlds + (ni0 * 16 + mfr) * LDST, kgrp);
            Frag b1 = load_b(Btlds + (ni1 * 16 + mfr) * LDST, kgrp);
            c0 = wmma_bf16(a, b0, c0);
            c1 = wmma_bf16(a, b1, c1);
        }
        #pragma unroll
        for (int r = 0; r < 8; ++r) {
            size_t mr = row0 + mi * 16 + r + (kgrp << 3);
            int cA = cg * 64 + ni0 * 16 + mfr;
            int cB = cg * 64 + ni1 * 16 + mfr;
            float vA = c0[r], vB = c1[r];
            if (mode == 1) {  // elu(x)+1 == x>0 ? x+1 : exp(x)
                vA = vA > 0.f ? vA + 1.f : __expf(vA);
                vB = vB > 0.f ? vB + 1.f : __expf(vB);
            }
            if (mode == 2) {
                ((float*)outp)[mr * DD + cA] = vA;
                ((float*)outp)[mr * DD + cB] = vB;
            } else {
                ((bf16*)outp)[mr * DD + cA] = (bf16)vA;
                ((bf16*)outp)[mr * DD + cB] = (bf16)vB;
            }
        }
    }
}

// ---------------------------------------------------------------------------
// KV kernel: per (b,h)  KV(64x64) = K_h^T @ V_h, plus Ksum(64).
// ---------------------------------------------------------------------------
__global__ __launch_bounds__(256) void kv_kernel(const bf16* __restrict__ Kf,
                                                 const bf16* __restrict__ Vf,
                                                 float* __restrict__ KVo,
                                                 float* __restrict__ Ksum) {
    __shared__ __align__(16) bf16 Alds[64 * LDST];
    __shared__ __align__(16) bf16 Btlds[64 * LDST];

    const int bh = blockIdx.x;
    const int b = bh >> 3, h = bh & 7;
    const int tid = threadIdx.x;
    const int lane = tid & 31;
    const int wave = tid >> 5;
    const int mi = wave >> 1;
    const int ni0 = (wave & 1) * 2, ni1 = ni0 + 1;
    const int mfr = lane & 15, kgrp = lane >> 4;

    const int kl = tid >> 3;
    const int e0 = (tid & 7) * 8;

    v8f c0 = {}, c1 = {};
    float ks_acc = 0.f;

    for (int nb = 0; nb < NN; nb += 32) {
        __syncthreads();
        const size_t rowbase = ((size_t)(b * NN + nb + kl)) * DD + h * HD;
        v8bf kv8 = *(const v8bf*)(Kf + rowbase + e0);
        v8bf vv8 = *(const v8bf*)(Vf + rowbase + e0);
        #pragma unroll
        for (int j = 0; j < 8; ++j) {
            Alds[(e0 + j) * LDST + kl]  = kv8[j];
            Btlds[(e0 + j) * LDST + kl] = vv8[j];
        }
        __syncthreads();
        if (tid < 64) {
            const bf16* ar = Alds + tid * LDST;
            #pragma unroll
            for (int j = 0; j < 32; ++j) ks_acc += (float)ar[j];
        }
        Frag a  = load_a(Alds + (mi * 16 + mfr) * LDST, kgrp);
        Frag b0 = load_b(Btlds + (ni0 * 16 + mfr) * LDST, kgrp);
        Frag b1 = load_b(Btlds + (ni1 * 16 + mfr) * LDST, kgrp);
        c0 = wmma_bf16(a, b0, c0);
        c1 = wmma_bf16(a, b1, c1);
    }

    float* kvdst = KVo + (size_t)bh * (HD * HD);
    #pragma unroll
    for (int r = 0; r < 8; ++r) {
        int d = mi * 16 + r + (kgrp << 3);
        kvdst[d * HD + ni0 * 16 + mfr] = c0[r];
        kvdst[d * HD + ni1 * 16 + mfr] = c1[r];
    }
    if (tid < 64) Ksum[bh * HD + tid] = ks_acc;
}

// ---------------------------------------------------------------------------
// Fused attention output (dominant kernel):
//   out[b,n,h*64+c] = (Q_h @ KV_h)[n,c] / clip(Q_h[n]·Ksum_h, 1e-6)
//                   + 0.1 * (P @ V_flat[b])[n, h*64+c]
// 128-row blocks, 2x2 C-tiles per wave (4 WMMA / 32-K step),
// double-buffered LDS with async global->LDS staging for the P tile.
// grid = (NN/128, H, B); block 256.
// ---------------------------------------------------------------------------
__global__ __launch_bounds__(256) void attn_kernel(const bf16* __restrict__ P,
                                                   const bf16* __restrict__ Vf,
                                                   const bf16* __restrict__ Qf,
                                                   const float* __restrict__ KV,
                                                   const float* __restrict__ Ksum,
                                                   bf16* __restrict__ outm) {
    __shared__ __align__(16) bf16 Alds[2][AM * LDST];    // P rows (128x32 per buf)
    __shared__ __align__(16) bf16 Btlds[2][64 * LDST];   // V cols, transposed
    __shared__ float normlds[AM];

    const int ntile = blockIdx.x, h = blockIdx.y, b = blockIdx.z;
    const int bh = b * HH + h;
    const size_t row0 = (size_t)ntile * AM;
    const int col0 = h * HD;

    const int tid = threadIdx.x;
    const int lane = tid & 31;
    const int wave = tid >> 5;
    const int mi0 = (wave >> 1) * 2, mi1 = mi0 + 1;     // 2 row-tiles (of 8)
    const int ni0 = (wave & 1) * 2,  ni1 = ni0 + 1;     // 2 col-tiles (of 4)
    const int mfr = lane & 15, kgrp = lane >> 4;

    // per-row normalizer
    if (tid < AM) {
        const bf16* q = Qf + ((size_t)b * NN + row0 + tid) * DD + col0;
        const float* s = Ksum + bh * HD;
        float acc = 0.f;
        #pragma unroll
        for (int d = 0; d < HD; ++d) acc += (float)q[d] * s[d];
        normlds[tid] = fmaxf(acc, 1e-6f);
    }

    const int am = tid >> 1;            // 0..127 (P row)
    const int ak = (tid & 1) * 16;      // 0 or 16 (k offset; 16 halfs = 32B)
    const int kl = tid >> 3;            // 0..31  (V row in k-block)
    const int e0 = (tid & 7) * 8;       // channel start

    // async staging of the P tile: 2 x b128 per thread (32 contiguous bytes)
    auto stageA = [&](int buf, int nb) {
        unsigned int lds = (unsigned int)(uintptr_t)(&Alds[buf][am * LDST + ak]);
        unsigned long long g =
            (unsigned long long)(uintptr_t)(P + (row0 + am) * (size_t)NN + nb + ak);
        asm volatile(
            "global_load_async_to_lds_b128 %0, %1, off\n\t"
            "global_load_async_to_lds_b128 %0, %1, off offset:16"
            :: "v"(lds), "v"(g) : "memory");
    };
    // synchronous transposed staging of the V tile
    auto stageB = [&](int buf, int nb) {
        v8bf vv8 = *(const v8bf*)(Vf + ((size_t)(b * NN + nb + kl)) * DD + col0 + e0);
        #pragma unroll
        for (int j = 0; j < 8; ++j) Btlds[buf][(e0 + j) * LDST + kl] = vv8[j];
    };

    // ----- dense graph-bias branch: K = 4096 in 128 steps, double-buffered -----
    v8f a00 = {}, a01 = {}, a10 = {}, a11 = {};
    stageA(0, 0);
    stageB(0, 0);
    asm volatile("s_wait_asynccnt 0" ::: "memory");
    __syncthreads();
    for (int s = 0; s < NN / 32; ++s) {
        const int cur = s & 1;
        if (s + 1 < NN / 32) {
            stageA(cur ^ 1, (s + 1) * 32);
            stageB(cur ^ 1, (s + 1) * 32);
        }
        Frag fa0 = load_a(&Alds[cur][(mi0 * 16 + mfr) * LDST], kgrp);
        Frag fa1 = load_a(&Alds[cur][(mi1 * 16 + mfr) * LDST], kgrp);
        Frag fb0 = load_b(&Btlds[cur][(ni0 * 16 + mfr) * LDST], kgrp);
        Frag fb1 = load_b(&Btlds[cur][(ni1 * 16 + mfr) * LDST], kgrp);
        a00 = wmma_bf16(fa0, fb0, a00);
        a01 = wmma_bf16(fa0, fb1, a01);
        a10 = wmma_bf16(fa1, fb0, a10);
        a11 = wmma_bf16(fa1, fb1, a11);
        asm volatile("s_wait_asynccnt 0" ::: "memory");
        __syncthreads();
    }

    // ----- linear branch: Q(128x64) @ KV(64x64), K = 64 in 2 steps -----
    v8f l00 = {}, l01 = {}, l10 = {}, l11 = {};
    for (int ks = 0; ks < 2; ++ks) {
        __syncthreads();
        {
            const bf16* qsrc = Qf + ((size_t)b * NN + row0 + am) * DD + col0 + ks * 32 + ak;
            *(v8bf*)(&Alds[0][am * LDST + ak])     = *(const v8bf*)(qsrc);
            *(v8bf*)(&Alds[0][am * LDST + ak + 8]) = *(const v8bf*)(qsrc + 8);
        }
        {
            const float* kvr = KV + (size_t)bh * (HD * HD) + (ks * 32 + kl) * HD + e0;
            #pragma unroll
            for (int j = 0; j < 8; ++j) Btlds[0][(e0 + j) * LDST + kl] = (bf16)kvr[j];
        }
        __syncthreads();
        Frag fa0 = load_a(&Alds[0][(mi0 * 16 + mfr) * LDST], kgrp);
        Frag fa1 = load_a(&Alds[0][(mi1 * 16 + mfr) * LDST], kgrp);
        Frag fb0 = load_b(&Btlds[0][(ni0 * 16 + mfr) * LDST], kgrp);
        Frag fb1 = load_b(&Btlds[0][(ni1 * 16 + mfr) * LDST], kgrp);
        l00 = wmma_bf16(fa0, fb0, l00);
        l01 = wmma_bf16(fa0, fb1, l01);
        l10 = wmma_bf16(fa1, fb0, l10);
        l11 = wmma_bf16(fa1, fb1, l11);
    }
    __syncthreads();

    // ----- combine + store -----
    #pragma unroll
    for (int r = 0; r < 8; ++r) {
        int mrA = mi0 * 16 + r + (kgrp << 3);
        int mrB = mi1 * 16 + r + (kgrp << 3);
        float invA = 1.0f / normlds[mrA];
        float invB = 1.0f / normlds[mrB];
        size_t baseA = ((size_t)b * NN + row0 + mrA) * DD + col0;
        size_t baseB = ((size_t)b * NN + row0 + mrB) * DD + col0;
        outm[baseA + ni0 * 16 + mfr] = (bf16)(l00[r] * invA + 0.1f * a00[r]);
        outm[baseA + ni1 * 16 + mfr] = (bf16)(l01[r] * invA + 0.1f * a01[r]);
        outm[baseB + ni0 * 16 + mfr] = (bf16)(l10[r] * invB + 0.1f * a10[r]);
        outm[baseB + ni1 * 16 + mfr] = (bf16)(l11[r] * invB + 0.1f * a11[r]);
    }
}

// ---------------------------------------------------------------------------
// host launch
// ---------------------------------------------------------------------------
extern "C" void kernel_launch(void* const* d_in, const int* in_sizes, int n_in,
                              void* d_out, int out_size, void* d_ws, size_t ws_size,
                              hipStream_t stream) {
    const float* x  = (const float*)d_in[0];
    const float* gb = (const float*)d_in[1];
    const float* wsrc[8] = {
        (const float*)d_in[2], (const float*)d_in[3],   // w_qd, w_qu
        (const float*)d_in[4], (const float*)d_in[5],   // w_kd, w_ku
        (const float*)d_in[6], (const float*)d_in[7],   // w_vd, w_vu
        (const float*)d_in[8], (const float*)d_in[9],   // w_od, w_ou
    };
    float* out = (float*)d_out;

    char* base = (char*)d_ws;
    size_t off = 0;
    auto alloc = [&](size_t bytes) -> char* {
        off = (off + 255) & ~(size_t)255;
        char* p = base + off;
        off += bytes;
        return p;
    };
    const size_t BND = (size_t)ROWS * DD;
    bf16* xh = (bf16*)alloc(BND * 2);
    bf16* P  = (bf16*)alloc((size_t)NN * NN * 2);
    bf16* Qf = (bf16*)alloc(BND * 2);
    bf16* Kf = (bf16*)alloc(BND * 2);
    bf16* Vf = (bf16*)alloc(BND * 2);
    bf16* Om = (bf16*)alloc(BND * 2);
    bf16* wh[8];
    for (int i = 0; i < 8; ++i) wh[i] = (bf16*)alloc((size_t)RK * DD * 2);
    float* KVb  = (float*)alloc((size_t)BB * HH * HD * HD * 4);
    float* Ksm  = (float*)alloc((size_t)BB * HH * HD * 4);
    (void)ws_size; (void)in_sizes; (void)n_in; (void)out_size;

    // 1. converts
    cvt_f32_bf16<<<(int)((BND + 255) / 256), 256, 0, stream>>>(x, xh, (int)BND);
    for (int i = 0; i < 8; ++i)
        cvt_f32_bf16<<<(RK * DD + 255) / 256, 256, 0, stream>>>(wsrc[i], wh[i], RK * DD);

    // 2. softmax of graph bias
    softmax_kernel<<<NN, 256, 0, stream>>>(gb, P);

    // 3. projections (Q,K with elu+1; V plain)
    proj_kernel<<<ROWS / 64, 256, 0, stream>>>(xh, wh[0], wh[1], Qf, 1);
    proj_kernel<<<ROWS / 64, 256, 0, stream>>>(xh, wh[2], wh[3], Kf, 1);
    proj_kernel<<<ROWS / 64, 256, 0, stream>>>(xh, wh[4], wh[5], Vf, 0);

    // 4. per-head KV + Ksum
    kv_kernel<<<BB * HH, 256, 0, stream>>>(Kf, Vf, KVb, Ksm);

    // 5. fused linear attention + graph-bias GEMM (async-staged, double-buffered)
    attn_kernel<<<dim3(NN / AM, HH, BB), 256, 0, stream>>>(P, Vf, Qf, KVb, Ksm, Om);

    // 6. final low-rank projection -> f32 output
    proj_kernel<<<ROWS / 64, 256, 0, stream>>>(Om, wh[6], wh[7], out, 2);
}
